// selfAttention_19774029431009
// MI455X (gfx1250) — matrix operand order repaired and checked by reading, not compiled, not run on hardware
//
#include <hip/hip_runtime.h>
#include <hip/hip_bf16.h>
#include <stdint.h>

// MI455X / gfx1250 flash-attention + output projection, WMMA f16 path.
// Round 4: register-transposed V staging (ds_store_b64 only), log2-domain
// online softmax (exp2, no per-exp v_mul), async mask tile, prefetch.
typedef __attribute__((ext_vector_type(16))) _Float16 v16h;
typedef __attribute__((ext_vector_type(8)))  _Float16 v8h;
typedef __attribute__((ext_vector_type(8)))  float    v8f;

#define N_B     2
#define L_SEQ   2048
#define E_DIM   1024
#define H_HEADS 16
#define D_HEAD  64

// scale * log2(e): softmax computed in log2 domain (ratios unchanged)
#define SCL_LOG2  (0.125f * 1.44269504088896340736f)
#define NEG_LOG2  (-1.8033688e19f)   // (-1e20) * 0.125 * log2(e)

__device__ __forceinline__ float rmax16(float v) {
#pragma unroll
  for (int m = 1; m < 16; m <<= 1) v = fmaxf(v, __shfl_xor(v, m, 32));
  return v;
}
__device__ __forceinline__ float rsum16(float v) {
#pragma unroll
  for (int m = 1; m < 16; m <<= 1) v += __shfl_xor(v, m, 32);
  return v;
}

union PackH4 { _Float16 h[4]; uint2 u; };

// One block = one (n, h, 64-query tile). 4 waves, each wave owns 16 query rows.
__global__ __launch_bounds__(128)
void flash_attn_kernel(const float* __restrict__ q,
                       const float* __restrict__ k,
                       const float* __restrict__ v,
                       const int*   __restrict__ mask,
                       _Float16*    __restrict__ ws)
{
  __shared__ _Float16 Ks[64][64];      // K tile, row = key, col = d   ( 8 KB)
  __shared__ _Float16 Vst[64][64];     // V tile transposed: [d][key]  ( 8 KB)
  __shared__ _Float16 Ps[4][16][64];   // per-wave P re-layout scratch ( 8 KB)
  __shared__ int      Msk[64][64];     // mask tile via async-to-LDS   (16 KB)

  const int tid  = threadIdx.x;
  const int w    = tid >> 5;
  const int lane = tid & 31;
  const int hs   = lane >> 4;   // half-wave select
  const int lq   = lane & 15;

  const int nh = blockIdx.x;
  const int n  = nh >> 4;
  const int h  = nh & 15;
  const int qb = blockIdx.y * 64;

  const size_t head_base = (size_t)n * L_SEQ * E_DIM + (size_t)h * D_HEAD;

  // ---- Q fragments in WMMA A-layout: two 16x32 f16 chunks over D=64 ----
  v16h aq[2];
  {
    const int qrow = qb + w * 16 + lq;          // A-layout: M = lane%16
    const float* qp = q + head_base + (size_t)qrow * E_DIM;
#pragma unroll
    for (int dc = 0; dc < 2; ++dc) {
#pragma unroll
      for (int i = 0; i < 16; ++i) {
        const int kk = dc * 32 + hs * 16 + i;
        aq[dc][i] = (_Float16)qp[kk];
      }
    }
  }

  v8f acc[4] = {};                  // O accumulator: 16 x 64 f32 per wave
  float mrun[8], lrun[8];
#pragma unroll
  for (int r = 0; r < 8; ++r) { mrun[r] = -3.0e38f; lrun[r] = 0.0f; }

  const float* kp = k + head_base;
  const float* vp = v + head_base;
  const size_t mbase = (size_t)n * L_SEQ * L_SEQ;

  for (int jb = 0; jb < L_SEQ / 64; ++jb) {
    const int kb = jb * 64;
    __syncthreads();

    // ---- async copy of the mask tile (64 q-rows x 64 keys, int32) ----
#pragma unroll
    for (int it = 0; it < 8; ++it) {
      const int c   = tid + it * 128;   // 0..1023, 16B chunks
      const int row = c >> 4;           // 0..63
      const int k4  = (c & 15) * 4;     // 0..60
      const int* gp = mask + mbase + (size_t)(qb + row) * L_SEQ + kb + k4;
      const unsigned loff = (unsigned)(uintptr_t)&Msk[row][k4];
      asm volatile("global_load_async_to_lds_b128 %0, %1, off"
                   :: "v"(loff), "v"(gp) : "memory");
    }

    // ---- prefetch next K/V tile while staging/computing this one ----
    if (jb + 1 < L_SEQ / 64) {
      const size_t nrow = (size_t)(kb + 64 + (tid >> 1)) * E_DIM + (tid & 1) * 32;
      __builtin_prefetch(kp + nrow, 0, 0);
      __builtin_prefetch(vp + nrow, 0, 0);
    }

    // ---- K tile staging, f32 -> f16, row-major, packed b64 stores ----
#pragma unroll
    for (int it = 0; it < 8; ++it) {
      const int e   = tid + it * 128;   // 1024 float4 slots
      const int row = e >> 4;           // 0..63 (key)
      const int c4  = (e & 15) * 4;     // 0..60 (d)
      const float4 kv = *(const float4*)(kp + (size_t)(kb + row) * E_DIM + c4);
      PackH4 pk;
      pk.h[0] = (_Float16)kv.x; pk.h[1] = (_Float16)kv.y;
      pk.h[2] = (_Float16)kv.z; pk.h[3] = (_Float16)kv.w;
      *(uint2*)&Ks[row][c4] = pk.u;
    }

    // ---- V tile staging: 4x4 register-block transpose, b64 stores ----
#pragma unroll
    for (int pass = 0; pass < 2; ++pass) {
      const int b  = tid + pass * 128;  // 0..255 4x4 blocks
      const int k0 = (b & 15) * 4;      // key base
      const int d0 = (b >> 4) * 4;      // d base
      const float4 r0 = *(const float4*)(vp + (size_t)(kb + k0 + 0) * E_DIM + d0);
      const float4 r1 = *(const float4*)(vp + (size_t)(kb + k0 + 1) * E_DIM + d0);
      const float4 r2 = *(const float4*)(vp + (size_t)(kb + k0 + 2) * E_DIM + d0);
      const float4 r3 = *(const float4*)(vp + (size_t)(kb + k0 + 3) * E_DIM + d0);
      PackH4 t;
      t.h[0] = (_Float16)r0.x; t.h[1] = (_Float16)r1.x;
      t.h[2] = (_Float16)r2.x; t.h[3] = (_Float16)r3.x;
      *(uint2*)&Vst[d0 + 0][k0] = t.u;
      t.h[0] = (_Float16)r0.y; t.h[1] = (_Float16)r1.y;
      t.h[2] = (_Float16)r2.y; t.h[3] = (_Float16)r3.y;
      *(uint2*)&Vst[d0 + 1][k0] = t.u;
      t.h[0] = (_Float16)r0.z; t.h[1] = (_Float16)r1.z;
      t.h[2] = (_Float16)r2.z; t.h[3] = (_Float16)r3.z;
      *(uint2*)&Vst[d0 + 2][k0] = t.u;
      t.h[0] = (_Float16)r0.w; t.h[1] = (_Float16)r1.w;
      t.h[2] = (_Float16)r2.w; t.h[3] = (_Float16)r3.w;
      *(uint2*)&Vst[d0 + 3][k0] = t.u;
    }

    asm volatile("s_wait_asynccnt 0x0" ::: "memory");
    __syncthreads();

    // ---- S = Q K^T : four 16x16 tiles, K reduced over D in 2 chunks ----
    v8f s[4] = {};
#pragma unroll
    for (int dc = 0; dc < 2; ++dc) {
#pragma unroll
      for (int t = 0; t < 4; ++t) {
        const v16h b = *(const v16h*)&Ks[t * 16 + lq][dc * 32 + hs * 16];
        s[t] = __builtin_amdgcn_wmma_f32_16x16x32_f16(false, aq[dc], false, b,
                                                      (short)0, s[t], false, false);
      }
    }

    // ---- mask (pre-scale, as in reference) + online softmax (log2) ----
#pragma unroll
    for (int r = 0; r < 8; ++r) {
      const int lrow = w * 16 + hs * 8 + r;   // row within the 64-row tile
      const int mk0 = Msk[lrow][lq];
      const int mk1 = Msk[lrow][16 + lq];
      const int mk2 = Msk[lrow][32 + lq];
      const int mk3 = Msk[lrow][48 + lq];
      const float e0 = mk0 ? s[0][r] * SCL_LOG2 : NEG_LOG2;
      const float e1 = mk1 ? s[1][r] * SCL_LOG2 : NEG_LOG2;
      const float e2 = mk2 ? s[2][r] * SCL_LOG2 : NEG_LOG2;
      const float e3 = mk3 ? s[3][r] * SCL_LOG2 : NEG_LOG2;
      // local 4-way max/sum, then a single 4-stage butterfly each
      const float mx   = rmax16(fmaxf(fmaxf(e0, e1), fmaxf(e2, e3)));
      const float mnew = fmaxf(mrun[r], mx);
      const float alpha = exp2f(mrun[r] - mnew);
      const float p0 = exp2f(e0 - mnew);
      const float p1 = exp2f(e1 - mnew);
      const float p2 = exp2f(e2 - mnew);
      const float p3 = exp2f(e3 - mnew);
      lrun[r] = alpha * lrun[r] + rsum16((p0 + p1) + (p2 + p3));
      mrun[r] = mnew;
#pragma unroll
      for (int ct = 0; ct < 4; ++ct) acc[ct][r] *= alpha;
      Ps[w][hs * 8 + r][lq]      = (_Float16)p0;
      Ps[w][hs * 8 + r][16 + lq] = (_Float16)p1;
      Ps[w][hs * 8 + r][32 + lq] = (_Float16)p2;
      Ps[w][hs * 8 + r][48 + lq] = (_Float16)p3;
    }

    // ---- P back in A-layout (two 16x32 chunks), O += P @ V ----
    v16h ap0, ap1;
    {
      const v8h lo0 = *(const v8h*)&Ps[w][lq][hs * 16];
      const v8h hi0 = *(const v8h*)&Ps[w][lq][8 + hs * 16];
      const v8h lo1 = *(const v8h*)&Ps[w][lq][32 + hs * 16];
      const v8h hi1 = *(const v8h*)&Ps[w][lq][40 + hs * 16];
#pragma unroll
      for (int i = 0; i < 8; ++i) {
        ap0[i] = lo0[i]; ap0[8 + i] = hi0[i];
        ap1[i] = lo1[i]; ap1[8 + i] = hi1[i];
      }
    }
#pragma unroll
    for (int ct = 0; ct < 4; ++ct) {
      const v16h bv0 = *(const v16h*)&Vst[ct * 16 + lq][hs * 16];
      acc[ct] = __builtin_amdgcn_wmma_f32_16x16x32_f16(false, ap0, false, bv0,
                                                       (short)0, acc[ct], false, false);
      const v16h bv1 = *(const v16h*)&Vst[ct * 16 + lq][32 + hs * 16];
      acc[ct] = __builtin_amdgcn_wmma_f32_16x16x32_f16(false, ap1, false, bv1,
                                                       (short)0, acc[ct], false, false);
    }
  }

  // ---- normalize, store f16 intermediate (N, L, E) to workspace ----
#pragma unroll
  for (int ct = 0; ct < 4; ++ct) {
#pragma unroll
    for (int r = 0; r < 8; ++r) {
      const int qrow = qb + w * 16 + hs * 8 + r;
      const int col  = h * D_HEAD + ct * 16 + lq;
      const float o  = acc[ct][r] / lrun[r];
      ws[(size_t)(n * L_SEQ + qrow) * E_DIM + col] = (_Float16)o;
    }
  }
}

// C(4096x1024) = Attn_f16 @ W^T + b.  64x64 C tile per block, 4 waves,
// 64-wide K chunks (8 WMMAs per stage, 16 stages).
__global__ __launch_bounds__(128)
void proj_kernel(const _Float16* __restrict__ A,
                 const float* __restrict__ Wt,     // (E, E) row-major: W[e, i]
                 const float* __restrict__ bias,
                 float* __restrict__ out)
{
  __shared__ _Float16 Ws[64][64];   // B chunk: [e_local][i_local] (8 KB)

  const int tid  = threadIdx.x;
  const int w    = tid >> 5;
  const int lane = tid & 31;
  const int hs   = lane >> 4;
  const int lq   = lane & 15;

  const int eb = blockIdx.x * 64;   // output-feature tile
  const int rb = blockIdx.y * 64;   // row tile

  v8f acc[4] = {};
  const int m = rb + w * 16 + lq;
  const _Float16* arow = A + (size_t)m * E_DIM;

  for (int kt = 0; kt < E_DIM / 64; ++kt) {
    __syncthreads();
#pragma unroll
    for (int it = 0; it < 8; ++it) {
      const int e  = tid + it * 128;   // 1024 float4 slots
      const int el = e >> 4;           // 0..63
      const int il = (e & 15) * 4;     // 0..60
      const float4 wv = *(const float4*)(Wt + (size_t)(eb + el) * E_DIM + kt * 64 + il);
      PackH4 pw;
      pw.h[0] = (_Float16)wv.x; pw.h[1] = (_Float16)wv.y;
      pw.h[2] = (_Float16)wv.z; pw.h[3] = (_Float16)wv.w;
      *(uint2*)&Ws[el][il] = pw.u;
    }
    __syncthreads();

    v16h a0, a1;
    {
      const v8h lo0 = *(const v8h*)(arow + kt * 64 + hs * 16);
      const v8h hi0 = *(const v8h*)(arow + kt * 64 + 8 + hs * 16);
      const v8h lo1 = *(const v8h*)(arow + kt * 64 + 32 + hs * 16);
      const v8h hi1 = *(const v8h*)(arow + kt * 64 + 40 + hs * 16);
#pragma unroll
      for (int i = 0; i < 8; ++i) {
        a0[i] = lo0[i]; a0[8 + i] = hi0[i];
        a1[i] = lo1[i]; a1[8 + i] = hi1[i];
      }
    }
#pragma unroll
    for (int ct = 0; ct < 4; ++ct) {
      const v16h b0 = *(const v16h*)&Ws[ct * 16 + lq][hs * 16];
      acc[ct] = __builtin_amdgcn_wmma_f32_16x16x32_f16(false, a0, false, b0,
                                                       (short)0, acc[ct], false, false);
      const v16h b1 = *(const v16h*)&Ws[ct * 16 + lq][32 + hs * 16];
      acc[ct] = __builtin_amdgcn_wmma_f32_16x16x32_f16(false, a1, false, b1,
                                                       (short)0, acc[ct], false, false);
    }
  }

#pragma unroll
  for (int ct = 0; ct < 4; ++ct) {
#pragma unroll
    for (int r = 0; r < 8; ++r) {
      const int row = rb + w * 16 + hs * 8 + r;
      const int e   = eb + ct * 16 + lq;
      out[(size_t)row * E_DIM + e] = acc[ct][r] + bias[e];
    }
  }
}

extern "C" void kernel_launch(void* const* d_in, const int* in_sizes, int n_in,
                              void* d_out, int out_size, void* d_ws, size_t ws_size,
                              hipStream_t stream) {
  (void)in_sizes; (void)n_in; (void)out_size; (void)ws_size;
  const float* values = (const float*)d_in[0];
  const float* key    = (const float*)d_in[1];
  const float* query  = (const float*)d_in[2];
  const int*   mask   = (const int*)d_in[3];
  const float* W_out  = (const float*)d_in[4];
  const float* b_out  = (const float*)d_in[5];
  _Float16* wsA = (_Float16*)d_ws;          // 8 MB f16 intermediate (N*L*E)
  float* out = (float*)d_out;

  flash_attn_kernel<<<dim3(N_B * H_HEADS, L_SEQ / 64), 128, 0, stream>>>(
      query, key, values, mask, wsA);
  proj_kernel<<<dim3(E_DIM / 64, (N_B * L_SEQ) / 64), 128, 0, stream>>>(
      wsA, W_out, b_out, out);
}
